// PathwayAttentionPooling_50895362458306
// MI455X (gfx1250) — matrix operand order repaired and checked by reading, not compiled, not run on hardware
//
#include <hip/hip_runtime.h>

#define H 128

typedef __attribute__((ext_vector_type(16))) _Float16 v16h;
typedef __attribute__((ext_vector_type(8)))  float    v8f;

// ---------------- zero accumulators ----------------
__global__ void k_zero(float* __restrict__ p, int n) {
    int i = blockIdx.x * blockDim.x + threadIdx.x;
    if (i < n) p[i] = 0.0f;
}

// ---------------- per-entry attention scores ----------------
// scores[e] = dot(protein_h[prot_idx[e]], w_attn)
__global__ void k_scores(const float* __restrict__ protein_h,
                         const float* __restrict__ w_attn,
                         const int*   __restrict__ prot_idx,
                         float*       __restrict__ scores,
                         int E) {
    __shared__ float wa[H];
    for (int i = threadIdx.x; i < H; i += blockDim.x) wa[i] = w_attn[i];
    __syncthreads();
    int e = blockIdx.x * blockDim.x + threadIdx.x;
    if (e >= E) return;
    const float* row = protein_h + (long)prot_idx[e] * H;
    float s = 0.0f;
#pragma unroll
    for (int h = 0; h < H; h += 4) {
        float4 v = *(const float4*)(row + h);
        s += v.x * wa[h] + v.y * wa[h + 1] + v.z * wa[h + 2] + v.w * wa[h + 3];
    }
    scores[e] = s;
}

// ---------------- per-group softmax + weighted pool (1 wave / group) ----------
__global__ void k_pool(const float* __restrict__ protein_h,
                       const float* __restrict__ scores,
                       const int*   __restrict__ prot_idx,
                       const int*   __restrict__ group_ids,
                       const int*   __restrict__ group_to_drug,
                       float*       __restrict__ drug_sum,
                       float*       __restrict__ drug_cnt,
                       int E, int G) {
    int wid  = (blockIdx.x * blockDim.x + threadIdx.x) >> 5;
    int lane = threadIdx.x & 31;
    if (wid >= G) return;
    int g = wid;

    // group_ids is sorted: binary search [start, end)
    int lo = 0, hi = E;
    while (lo < hi) { int mid = (lo + hi) >> 1; if (group_ids[mid] < g) lo = mid + 1; else hi = mid; }
    int start = lo;
    hi = E;
    while (lo < hi) { int mid = (lo + hi) >> 1; if (group_ids[mid] < g + 1) lo = mid + 1; else hi = mid; }
    int end = lo;
    if (start >= end) return;   // empty group contributes nothing

    // stable softmax (uniform scalar scan; avg segment length ~10)
    float m = -3.402823466e38f;
    for (int e = start; e < end; ++e) m = fmaxf(m, scores[e]);
    float denom = 0.0f;
    for (int e = start; e < end; ++e) denom += __expf(scores[e] - m);
    float inv = 1.0f / denom;

    // each lane owns 4 of the 128 h-dims -> 512B coalesced row loads
    int hb = lane * 4;
    float4 acc = make_float4(0.f, 0.f, 0.f, 0.f);
    for (int e = start; e < end; ++e) {
        float w = __expf(scores[e] - m) * inv;
        const float* row = protein_h + (long)prot_idx[e] * H;
        float4 v = *(const float4*)(row + hb);
        acc.x += w * v.x; acc.y += w * v.y; acc.z += w * v.z; acc.w += w * v.w;
    }

    int d = group_to_drug[g];
    float* dst = drug_sum + (long)d * H + hb;
    atomicAdd(dst + 0, acc.x);
    atomicAdd(dst + 1, acc.y);
    atomicAdd(dst + 2, acc.z);
    atomicAdd(dst + 3, acc.w);
    if (lane == 0) atomicAdd(drug_cnt + d, 1.0f);
}

// ---------------- fingerprint = mean over non-empty pools, cast to f16 -------
__global__ void k_finalize(const float* __restrict__ drug_sum,
                           const float* __restrict__ drug_cnt,
                           _Float16*    __restrict__ fp16,
                           int n) {
    int i = blockIdx.x * blockDim.x + threadIdx.x;
    if (i >= n) return;
    float c = drug_cnt[i >> 7];                 // i / H
    float v = (c > 0.0f) ? drug_sum[i] / c : 0.0f;
    fp16[i] = (_Float16)v;
}

__global__ void k_cvt_w(const float* __restrict__ w_out, _Float16* __restrict__ w16) {
    int i = blockIdx.x * blockDim.x + threadIdx.x;
    if (i < H * H) w16[i] = (_Float16)w_out[i];
}

// ---------------- out = relu(fp @ w_out^T + b_out) via WMMA ------------------
// One wave computes one 16x16 output tile; K = 128 = 4 x (16x16x32 f16 WMMA).
// M is a multiple of 16 (out_size / H), so the epilogue stores unconditionally.
__global__ void k_out_wmma(const _Float16* __restrict__ A16,  // [M,128] fingerprint f16
                           const _Float16* __restrict__ W16,  // [128,128] w_out (row j, col h)
                           const float*    __restrict__ b_out,
                           float*          __restrict__ out,
                           int mtiles) {
    int wid  = (blockIdx.x * blockDim.x + threadIdx.x) >> 5;
    int lane = threadIdx.x & 31;
    const int NT = H / 16;                       // 8 tiles along N
    int mt = wid / NT;
    int nt = wid % NT;
    if (mt >= mtiles) return;

    int lh  = lane & 15;                         // row (A) / col (B,C) within tile
    int hi2 = lane >> 4;                         // half-select per ISA layouts
    v8f c = {};

#pragma unroll
    for (int kb = 0; kb < H; kb += 32) {
        v16h a, b;
#pragma unroll
        for (int i = 0; i < 8; ++i) {
            // A (16x32 f16) layout: lanes0-15 VGPR0..3:K=0..7, VGPR4..7:K=16..23;
            //                       lanes16-31 shift by +8.  (8 contiguous halves
            //                       per 4-VGPR span -> compiler emits b128 loads)
            int ka = kb + ((i < 4) ? 0 : 16) + 8 * hi2 + 2 * (i & 3);
            const _Float16* ap = A16 + (long)(mt * 16 + lh) * H + ka;
            a[2 * i]     = ap[0];
            a[2 * i + 1] = ap[1];
            // B (32x16 f16): N = lh, K in same order as A (K=0..7,16..23; +8 for hi2).
            int kk = kb + ((i < 4) ? 0 : 16) + 8 * hi2 + 2 * (i & 3);
            const _Float16* bp = W16 + (long)(nt * 16 + lh) * H + kk;  // B[k][n] = w_out[n][k]
            b[2 * i]     = bp[0];
            b[2 * i + 1] = bp[1];
        }
        c = __builtin_amdgcn_wmma_f32_16x16x32_f16(
                /*neg_a=*/false, a, /*neg_b=*/false, b,
                /*c_mod=*/(short)0, c, /*reuse_a=*/false, /*reuse_b=*/false);
    }

    // C/D layout: VGPR j -> M = j + 8*hi2, N = lh. Straight-line epilogue.
    float bias = b_out[nt * 16 + lh];
    float* obase = out + (long)(mt * 16 + 8 * hi2) * H + nt * 16 + lh;
#pragma unroll
    for (int j = 0; j < 8; ++j) {
        float v = c[j] + bias;
        obase[(long)j * H] = fmaxf(v, 0.0f);
    }
}

extern "C" void kernel_launch(void* const* d_in, const int* in_sizes, int n_in,
                              void* d_out, int out_size, void* d_ws, size_t ws_size,
                              hipStream_t stream) {
    const float* protein_h     = (const float*)d_in[0];
    const float* w_attn        = (const float*)d_in[1];
    const float* w_out         = (const float*)d_in[2];
    const float* b_out         = (const float*)d_in[3];
    const int*   prot_idx      = (const int*)d_in[4];
    const int*   group_ids     = (const int*)d_in[5];
    const int*   group_to_drug = (const int*)d_in[6];

    int E  = in_sizes[4];
    int G  = in_sizes[6];
    int ND = out_size / H;          // number of drugs (multiple of 16)
    float* outp = (float*)d_out;

    // workspace layout
    char* ws = (char*)d_ws;
    size_t off = 0;
    float* scores = (float*)(ws + off); off += (size_t)E * sizeof(float);
    off = (off + 255) & ~(size_t)255;
    float* drug_sum = (float*)(ws + off); off += (size_t)ND * H * sizeof(float);
    float* drug_cnt = (float*)(ws + off); off += (size_t)ND * sizeof(float);
    off = (off + 255) & ~(size_t)255;
    _Float16* fp16 = (_Float16*)(ws + off); off += (size_t)ND * H * sizeof(_Float16);
    _Float16* w16  = (_Float16*)(ws + off); off += (size_t)H * H * sizeof(_Float16);
    (void)ws_size; (void)n_in;

    // 1) zero accumulators (drug_sum and drug_cnt are contiguous)
    int zn = ND * H + ND;
    k_zero<<<(zn + 255) / 256, 256, 0, stream>>>(drug_sum, zn);

    // 2) per-entry scores
    k_scores<<<(E + 255) / 256, 256, 0, stream>>>(protein_h, w_attn, prot_idx, scores, E);

    // 3) per-group softmax + pool (1 wave per group, 4 waves per block)
    int pool_blocks = (G + 3) / 4;
    k_pool<<<pool_blocks, 128, 0, stream>>>(protein_h, scores, prot_idx, group_ids,
                                            group_to_drug, drug_sum, drug_cnt, E, G);

    // 4) finalize fingerprint (f16) + convert w_out (f16)
    k_finalize<<<(ND * H + 255) / 256, 256, 0, stream>>>(drug_sum, drug_cnt, fp16, ND * H);
    k_cvt_w<<<(H * H + 255) / 256, 256, 0, stream>>>(w_out, w16);

    // 5) output projection + bias + relu via WMMA (1 wave per 16x16 tile)
    int mtiles = ND / 16;
    int tiles = mtiles * (H / 16);
    int gemm_blocks = (tiles + 7) / 8;            // 8 waves per 256-thread block
    k_out_wmma<<<gemm_blocks, 256, 0, stream>>>(fp16, w16, b_out, outp, mtiles);
}
